// SpanNeighborBuilder_86371792323239
// MI455X (gfx1250) — matrix operand order repaired and checked by reading, not compiled, not run on hardware
//
#include <hip/hip_runtime.h>
#include <hip/hip_bf16.h>

// ---------------- problem constants ----------------
#define BB   8
#define LL   128
#define SS   1928       // number of spans (width<=16 over L=128)
#define SP   2048       // S padded so M tiles (128) split into 16 blocks of 8 waves
#define KTOP 10
#define MT   (SP / 16)  // 128 row tiles
#define MBLK (MT / 8)   // 16 row-tile blocks (8 waves per block)
#define NG2  (SP / 64)  // 32 N-groups for GEMM2 (4 tiles of 16 per group)
#define NG1  (LL / 64)  // 2  N-groups for GEMM1

typedef _Float16 v16h __attribute__((ext_vector_type(16)));
typedef _Float16 v8h  __attribute__((ext_vector_type(8)));
typedef float    v8f  __attribute__((ext_vector_type(8)));

// ---------------- span enumeration helpers ----------------
__device__ __forceinline__ void span_lr(int s, int& l, int& r) {
  if (s < 1808) { l = s >> 4; r = l + (s & 15); }
  else {
    int rem = s - 1808; l = 113; int width = 15;
    while (rem >= width) { rem -= width; ++l; --width; }
    r = l + rem;
  }
}
__device__ __forceinline__ int span_id(int l, int r) {
  if (l < 0 || r < 0 || l > r || r >= LL || (r - l) >= 16) return -1;
  if (l <= 112) return (l << 4) + (r - l);
  int a = 128 - l;                       // 1..15
  return 1928 - ((a * (a + 1)) >> 1) + (r - l);
}

// ---------------- phase 0: G^T (f16) and per-(b,l) adjacency bitmasks ----------------
__global__ void prep_kernel(const float* __restrict__ dist,
                            _Float16* __restrict__ GT,      // [B][m][l] = exp(-dist[b,l,m])
                            unsigned* __restrict__ Abits) { // [B][l][4] bit m: dist<=2 || l==m
  int bl = blockIdx.x;            // b*L + l
  int b  = bl >> 7, l = bl & 127;
  int m  = threadIdx.x;           // 128 threads
  __shared__ unsigned w[4];
  if (m < 4) w[m] = 0u;
  __syncthreads();
  float d = dist[(size_t)bl * LL + m];
  float g = __expf(-d);           // GAMMA = 1
  GT[((size_t)b * LL + m) * LL + l] = (_Float16)g;
  if (d <= 2.0f || l == m) atomicOr(&w[m >> 5], 1u << (m & 31));
  __syncthreads();
  if (m < 4) Abits[(size_t)bl * 4 + m] = w[m];
}

// ---------------- phase 1: dense f16 P_w rows + Umask + top1 ----------------
__global__ void pack_kernel(const int*   __restrict__ hidx,  // [B,L,L,4]
                            const float* __restrict__ hw,    // [B,L,L,4]
                            const unsigned* __restrict__ Abits,
                            _Float16* __restrict__ P16,      // [B,SP,128]
                            int* __restrict__ top1,          // [B,SP]
                            int* __restrict__ meta,          // [B,SP] (init -1)
                            unsigned* __restrict__ Umask) {  // [B,SP,4]
  int bs = blockIdx.x;            // b*SP + s
  int b  = bs / SP, s = bs % SP;
  int j  = threadIdx.x;           // 128 threads
  if (s >= SS) {
    P16[(size_t)bs * LL + j] = (_Float16)0.0f;
    if (j == 0) { top1[bs] = -1; meta[bs] = -1; }
    if (j < 4) Umask[(size_t)bs * 4 + j] = 0u;
    return;
  }
  int l, r; span_lr(s, l, r);
  size_t base = (((size_t)b * LL + l) * LL + r) * 4;   // KTOK_TOTAL = 4, use first 3
  int   i0 = hidx[base + 0], i1 = hidx[base + 1], i2 = hidx[base + 2];
  float w0 = hw[base + 0],  w1 = hw[base + 1],  w2 = hw[base + 2];
  float sum = 0.0f;
  if (i0 >= 0 && i0 == j) sum += w0;
  if (i1 >= 0 && i1 == j) sum += w1;
  if (i2 >= 0 && i2 == j) sum += w2;
  P16[(size_t)bs * LL + j] = (_Float16)sum;
  if (j == 0) { top1[bs] = i0; meta[bs] = -1; }
  if (j < 4) {
    unsigned um = 0u;
    if (i0 >= 0) um |= Abits[((size_t)b * LL + i0) * 4 + j];
    if (i1 >= 0) um |= Abits[((size_t)b * LL + i1) * 4 + j];
    if (i2 >= 0) um |= Abits[((size_t)b * LL + i2) * 4 + j];
    Umask[(size_t)bs * 4 + j] = um;
  }
}

// ---------------- phase 2: cap counting (sequential per (b,l), disjoint writes) ------
__global__ void capk_kernel(const int* __restrict__ top1, int* __restrict__ meta) {
  int bl = blockIdx.x * blockDim.x + threadIdx.x;
  if (bl >= BB * LL) return;
  int b = bl >> 7, l = bl & 127;
  const int* t1 = top1 + (size_t)b * SP;
  int*       mt = meta + (size_t)b * SP;
  int cnt = 0;
  for (int t = 0; t < SS; ++t) {
    int v = t1[t];
    if (v >= 0 && ((v < LL - 1) ? v : (LL - 1)) == l) {
      ++cnt;
      if (cnt <= 128) mt[t] = l;   // CAP = 128
    }
  }
}

// ---------------- WMMA GEMM block: 8 waves x (16x64) output, K=128 -------------------
// A row-major MxK; Brows row-major NxK (= B transposed). The 64-row B panel for one
// N-group is a contiguous 16KB chunk (ld == K == 128 halves) -> async-copy it to LDS
// once per block, then all 8 waves read fragments from LDS.
template <bool OUT_F16>
__device__ __forceinline__ void gemm_block(const _Float16* __restrict__ A,
                                           const _Float16* __restrict__ Brows,
                                           void* __restrict__ Cptr, int ldC,
                                           int mb, int ng,
                                           _Float16* smemB /* LDS offset 0 */) {
  const int tid  = threadIdx.x;
  const int wave = tid >> 5;
  const int lane = tid & 31;
  const int hsel = lane >> 4;
  const int mr   = lane & 15;

  // ---- stage 64x128 f16 B panel (16KB, contiguous) into LDS, async, VGPR-bypassing.
  // smemB's address is passed into each asm (unused in the template) so the compiler
  // knows the async ops may write the LDS array; otherwise GlobalOpt proves the
  // shared array is never stored and folds every read of it to undef.
  {
    const char* src = (const char*)(Brows + (size_t)(ng * 64) * LL);
    _Float16*   lp  = smemB;
#pragma unroll
    for (int i = 0; i < 4; ++i) {
      unsigned           off = (unsigned)((i * 256 + tid) * 16);
      unsigned           dst = off;  // smemB is the only __shared__ object -> LDS offset 0
      unsigned long long gp  = (unsigned long long)(uintptr_t)(src + off);
      asm volatile("global_load_async_to_lds_b128 %0, %1, off"
                   :: "v"(dst), "v"(gp), "v"(lp) : "memory");
    }
    asm volatile("s_wait_asynccnt 0x0" ::: "memory");
  }
  __syncthreads();
  // Opaque point: the LDS contents were produced outside the compiler's view.
  { _Float16* lp = smemB; asm volatile("" : "+v"(lp) :: "memory"); }

  // ---- A fragments for this wave's M tile, all K chunks, held in registers
  const int tm = mb * 8 + wave;
  const _Float16* Arow = A + (size_t)(tm * 16 + mr) * LL;
  union AF { v16h v; v8h h2[2]; } afr[4];
#pragma unroll
  for (int kk4 = 0; kk4 < 4; ++kk4) {
    int ko = kk4 * 32 + hsel * 16;
    afr[kk4].h2[0] = *reinterpret_cast<const v8h*>(Arow + ko);
    afr[kk4].h2[1] = *reinterpret_cast<const v8h*>(Arow + ko + 8);
  }

  // ---- 4 independent accumulator chains (better XDL pipelining), B from LDS
  v8f acc[4] = {{}, {}, {}, {}};
#pragma unroll
  for (int kk4 = 0; kk4 < 4; ++kk4) {
#pragma unroll
    for (int j = 0; j < 4; ++j) {
      const _Float16* Brow = smemB + (size_t)(j * 16 + mr) * LL;
      v16h bf = *reinterpret_cast<const v16h*>(Brow + kk4 * 32 + hsel * 16);
      acc[j] = __builtin_amdgcn_wmma_f32_16x16x32_f16(false, afr[kk4].v, false, bf,
                                                      (short)0, acc[j], false, false);
    }
  }

  // ---- store 16x64 tile
#pragma unroll
  for (int j = 0; j < 4; ++j) {
    int colBase = (ng * 4 + j) * 16 + mr;
#pragma unroll
    for (int v = 0; v < 8; ++v) {
      size_t idx = (size_t)(tm * 16 + v + hsel * 8) * ldC + colBase;
      if (OUT_F16) ((_Float16*)Cptr)[idx] = (_Float16)acc[j][v];
      else         ((float*)Cptr)[idx]    = acc[j][v];
    }
  }
}

// GEMM1: A_pw[b] = P_w[b] (SPx128) x G[b] (128x128), f16 out. All batches in one grid.
__global__ void __launch_bounds__(256)
gemm1_kernel(const _Float16* __restrict__ P16,
             const _Float16* __restrict__ GT,
             _Float16* __restrict__ APW) {
  __shared__ __align__(64) _Float16 smemB[64 * LL];
  int bx  = blockIdx.x;                 // BB * MBLK * NG1 blocks
  int b   = bx / (MBLK * NG1);
  int rem = bx % (MBLK * NG1);
  int mb  = rem / NG1, ng = rem % NG1;
  const _Float16* A  = P16 + (size_t)b * SP * LL;
  const _Float16* Br = GT  + (size_t)b * LL * LL;   // rows = columns of G
  _Float16*       C  = APW + (size_t)b * SP * LL;
  gemm_block<true>(A, Br, C, LL, mb, ng, smemB);
}

// GEMM2 (one batch): Gate = A_pw (SPx128) x P_w^T (128xSP), f32 out.
__global__ void __launch_bounds__(256)
gemm2_kernel(const _Float16* __restrict__ APWb,
             const _Float16* __restrict__ P16b,
             float* __restrict__ Gate) {
  __shared__ __align__(64) _Float16 smemB[64 * LL];
  int bx = blockIdx.x;                  // MBLK * NG2 blocks
  int mb = bx / NG2, ng = bx % NG2;
  gemm_block<false>(APWb, P16b, Gate, SP, mb, ng, smemB);
}

// ---------------- phase 4: masked top-10, one wave per span row ----------------
__global__ void topk_kernel(const float* __restrict__ Gate,       // [SP,SP] batch b
                            const int* __restrict__ meta_b,       // [SP]
                            const unsigned* __restrict__ Umask_b, // [SP,4]
                            int* __restrict__ outIdx_b,           // [SS,K]
                            int* __restrict__ outMask_b) {        // [SS,K]
  int wid = (blockIdx.x * blockDim.x + threadIdx.x) >> 5;   // wave id == span s
  if (wid >= SS) return;
  const int s    = wid;
  const int lane = threadIdx.x & 31;

  int l, r; span_lr(s, l, r);
  const int dl[8] = {-1, -1, -1, 0, 0, 1, 1, 1};
  const int dr[8] = {-1,  0,  1, -1, 1, -1, 0, 1};
  int gid[8];
#pragma unroll
  for (int i = 0; i < 8; ++i) gid[i] = span_id(l + dl[i], r + dr[i]);

  unsigned um[4];
#pragma unroll
  for (int i = 0; i < 4; ++i) um[i] = Umask_b[(size_t)s * 4 + i];

  const float NEG = -__builtin_huge_valf();
  float bv[KTOP]; int bi[KTOP];
#pragma unroll
  for (int i = 0; i < KTOP; ++i) { bv[i] = NEG; bi[i] = s; }

  const float* grow = Gate + (size_t)s * SP;
  for (int t = lane; t < SS; t += 32) {
    int  m  = meta_b[t];
    bool ok = false;
    if (m >= 0 && t != s) ok = (um[m >> 5] >> (m & 31)) & 1u;
    if (!ok) {
#pragma unroll
      for (int i = 0; i < 8; ++i) ok = ok || (t == gid[i]);
    }
    if (!ok) continue;
    float v = grow[t];
    if (v > bv[KTOP - 1]) {                 // strict > keeps earlier-t on ties
      int p = KTOP - 1;
      while (p > 0 && v > bv[p - 1]) { bv[p] = bv[p - 1]; bi[p] = bi[p - 1]; --p; }
      bv[p] = v; bi[p] = t;
    }
  }

  // 10-round butterfly tournament over per-lane sorted lists
  int ptr = 0;
  for (int k = 0; k < KTOP; ++k) {
    float cv = (ptr < KTOP) ? bv[ptr] : NEG;
    int   ci = (ptr < KTOP) ? bi[ptr] : s;
    float mv = cv; int mi = ci; int ml = lane;
    for (int off = 16; off > 0; off >>= 1) {
      float ov = __shfl_xor(mv, off, 32);
      int   oi = __shfl_xor(mi, off, 32);
      int   ol = __shfl_xor(ml, off, 32);
      if (ov > mv || (ov == mv && oi < mi)) { mv = ov; mi = oi; ml = ol; }
    }
    if (lane == ml) ++ptr;                  // winner consumes its head
    if (lane == 0) {
      outIdx_b[(size_t)s * KTOP + k]  = mi; // mi==s already when masked (-inf sentinel)
      outMask_b[(size_t)s * KTOP + k] = 1;  // reference returns all-ones mask
    }
  }
}

// ---------------- host: workspace layout + launches ----------------
extern "C" void kernel_launch(void* const* d_in, const int* in_sizes, int n_in,
                              void* d_out, int out_size, void* d_ws, size_t ws_size,
                              hipStream_t stream) {
  (void)in_sizes; (void)n_in; (void)out_size; (void)ws_size;
  const int*   hidx = (const int*)  d_in[0];   // head_indices int32 [B,L,L,4]
  const float* hw   = (const float*)d_in[1];   // head_weights f32   [B,L,L,4]
  const float* dist = (const float*)d_in[2];   // dist f32           [B,L,L]

  char* ws = (char*)d_ws;
  size_t off = 0;
  auto take = [&](size_t bytes) { size_t o = off; off = (off + bytes + 255) & ~(size_t)255; return o; };
  _Float16* P16   = (_Float16*)(ws + take((size_t)BB * SP * LL * 2)); // 4.2 MB
  _Float16* APW   = (_Float16*)(ws + take((size_t)BB * SP * LL * 2)); // 4.2 MB
  _Float16* GT    = (_Float16*)(ws + take((size_t)BB * LL * LL * 2)); // 256 KB
  unsigned* Abits = (unsigned*)(ws + take((size_t)BB * LL * 4 * 4));  // 16 KB
  unsigned* Umask = (unsigned*)(ws + take((size_t)BB * SP * 4 * 4));  // 256 KB
  int*      top1  = (int*)     (ws + take((size_t)BB * SP * 4));
  int*      meta  = (int*)     (ws + take((size_t)BB * SP * 4));
  float*    Gate  = (float*)   (ws + take((size_t)SP * SP * 4));      // 16.8 MB, reused per batch

  prep_kernel<<<BB * LL, 128, 0, stream>>>(dist, GT, Abits);
  pack_kernel<<<BB * SP, 128, 0, stream>>>(hidx, hw, Abits, P16, top1, meta, Umask);
  capk_kernel<<<(BB * LL + 127) / 128, 128, 0, stream>>>(top1, meta);
  gemm1_kernel<<<BB * MBLK * NG1, 256, 0, stream>>>(P16, GT, APW);

  int* outIdx  = (int*)d_out;
  int* outMask = outIdx + (size_t)BB * SS * KTOP;
  for (int b = 0; b < BB; ++b) {
    gemm2_kernel<<<MBLK * NG2, 256, 0, stream>>>(
        APW + (size_t)b * SP * LL, P16 + (size_t)b * SP * LL, Gate);
    topk_kernel<<<(SS + 7) / 8, 256, 0, stream>>>(
        Gate, meta + (size_t)b * SP, Umask + (size_t)b * SP * 4,
        outIdx + (size_t)b * SS * KTOP, outMask + (size_t)b * SS * KTOP);
  }
}
